// Decoder_52432960749716
// MI455X (gfx1250) — matrix operand order repaired and checked by reading, not compiled, not run on hardware
//
#include <hip/hip_runtime.h>
#include <hip/hip_bf16.h>

// ---------------------------------------------------------------------------
// Decoder: LayerNormLSTM scan (B=2048, T=28, DIN=4, U=512) + Dense(4, relu)
// Batch-parallel decomposition: 128 WGs x 16 batch rows, full scan per WG.
// Recurrent GEMM done with V_WMMA_F32_16X16X32_BF16, B prepacked in d_ws.
// ---------------------------------------------------------------------------

typedef __attribute__((ext_vector_type(16))) __bf16 v16bf;
typedef __attribute__((ext_vector_type(8)))  float  v8f;

#define TSZ 28
#define USZ 512
#define NG  2048      // 4U gate columns
#define ROWS 16       // batch rows per workgroup
#define EPS 1e-3f

union ABfrag { unsigned u[8]; uint4 q[2]; v16bf v; };

__device__ __forceinline__ unsigned f2bf1(float f) {
  unsigned u = __float_as_uint(f);
  unsigned r = u + 0x7FFFu + ((u >> 16) & 1u);   // round-to-nearest-even
  return r >> 16;
}
__device__ __forceinline__ unsigned packbf(float lo, float hi) {
  return f2bf1(lo) | (f2bf1(hi) << 16);
}
__device__ __forceinline__ float sigf(float x) { return 1.0f / (1.0f + __expf(-x)); }
__device__ __forceinline__ float red16(float v) {
  // butterfly within 16-lane half (masks < 16 never cross halves of a wave32)
  v += __shfl_xor(v, 1, 32);
  v += __shfl_xor(v, 2, 32);
  v += __shfl_xor(v, 4, 32);
  v += __shfl_xor(v, 8, 32);
  return v;
}

// ---------------------------------------------------------------------------
// Prep: pack rec_kernel [512 x 2048] f32 -> bf16 WMMA B-fragments in d_ws.
// Fragment f = kt*128 + ntg  (kt: K-tile of 32, ntg: N-tile of 16).
// Per (f, lane): 8 dwords stored as two b128-coalesced groups of 4:
//   dword index = ((f*2 + (v>>2))*32 + lane)*4 + (v&3)
// Lane layout (contiguous k order used here, 16-bit B, K=32): n = lane&15, hi = lane>>4,
//   dword v holds K = hi*16 + 2v (low 16b) and K = hi*16 + 2v + 1 (high 16b).
// ---------------------------------------------------------------------------
__global__ void prep_pack_B(const float* __restrict__ rk, unsigned* __restrict__ out) {
  int idx = blockIdx.x * 256 + threadIdx.x;   // 524288 dwords total (2 MB)
  int dw   = idx & 3;
  int lane = (idx >> 2) & 31;
  int hf   = (idx >> 7) & 1;
  int f    = idx >> 8;                        // 0..2047
  int kt   = f >> 7;                          // 0..15
  int ntg  = f & 127;                         // 0..127
  int v    = hf * 4 + dw;                     // 0..7
  int hi   = lane >> 4, n = lane & 15;
  int k0   = kt * 32 + hi * 16 + 2 * v;
  int col  = ntg * 16 + n;
  out[idx] = packbf(rk[k0 * NG + col], rk[(k0 + 1) * NG + col]);
}

// ---------------------------------------------------------------------------
// Main persistent-scan kernel: one WG = 16 batch rows, all 28 steps.
// ---------------------------------------------------------------------------
__global__ __launch_bounds__(256)
void decoder_scan(const float* __restrict__ x, const float* __restrict__ h0,
                  const float* __restrict__ c0, const float* __restrict__ kern,
                  const float* __restrict__ bias,
                  const float* __restrict__ kn_g, const float* __restrict__ kn_b,
                  const float* __restrict__ rn_g, const float* __restrict__ rn_b,
                  const float* __restrict__ sn_g, const float* __restrict__ sn_b,
                  const float* __restrict__ Wd,  const float* __restrict__ bd,
                  const uint4* __restrict__ bfrag, float* __restrict__ out) {
  extern __shared__ char smem[];
  float*    kernL = (float*)smem;                 // 4*2048          = 8192 f
  float*    zbuf  = kernL + 4 * NG;               // 16*2052 (pad 4) = 32832 f
  float*    hL    = zbuf + 16 * 2052;             // 16*516          = 8256 f
  float*    cL    = hL + 16 * 516;                // 16*516          = 8256 f
  unsigned* hbf   = (unsigned*)(cL + 16 * 516);   // 16*260 dwords (bf16 pairs)
  float*    xr    = (float*)(hbf + 16 * 260);     // 16*4
  float*    zred  = xr + 64;                      // 16*2
  float*    yred  = zred + 32;                    // 16*2
  float*    cred  = yred + 32;                    // 16*2

  const int tid    = threadIdx.x;
  const int b0     = blockIdx.x * ROWS;
  const int waveId = tid >> 5;
  const int lane   = tid & 31;
  const int l      = lane & 15;
  const int hi     = lane >> 4;
  const int mrow   = tid >> 4;        // row mapping for LN / gate phases

  // ---- init: stage kernel weights, h0/c0, bf16-packed h ----
  for (int i = tid; i < 4 * NG; i += 256) kernL[i] = kern[i];
  for (int i = tid; i < ROWS * USZ; i += 256) {
    int m = i >> 9, u = i & 511;
    hL[m * 516 + u] = h0[(b0 + m) * USZ + u];
    cL[m * 516 + u] = c0[(b0 + m) * USZ + u];
  }
  __syncthreads();
  for (int i = tid; i < ROWS * 256; i += 256) {
    int m = i >> 8, kd = i & 255;
    hbf[m * 260 + kd] = packbf(hL[m * 516 + 2 * kd], hL[m * 516 + 2 * kd + 1]);
  }
  __syncthreads();

  const float4 bdv = *(const float4*)bd;

  for (int t = 0; t < TSZ; ++t) {
    if (tid < 32) { zred[tid] = 0.f; yred[tid] = 0.f; cred[tid] = 0.f; }
    if (tid < 64) xr[tid] = x[((b0 + (tid >> 2)) * TSZ + t) * 4 + (tid & 3)];
    __syncthreads();

    // ---- Phase A: zx = x_t @ kernel (K=4, trivial VALU) + row stats ----
    {
      float x0 = xr[mrow * 4 + 0], x1 = xr[mrow * 4 + 1];
      float x2 = xr[mrow * 4 + 2], x3 = xr[mrow * 4 + 3];
      float s = 0.f, sq = 0.f;
      for (int j = 0; j < 128; ++j) {
        int n = l + (j << 4);
        float zr = x0 * kernL[n] + x1 * kernL[NG + n] + x2 * kernL[2 * NG + n] +
                   x3 * kernL[3 * NG + n];
        zbuf[mrow * 2052 + n] = zr;
        s += zr; sq += zr * zr;
      }
      s = red16(s); sq = red16(sq);
      if (l == 0) { zred[mrow * 2] = s; zred[mrow * 2 + 1] = sq; }
    }
    __syncthreads();
    {   // LN(zx) with kn gains
      float mu = zred[mrow * 2] * (1.f / NG);
      float var = zred[mrow * 2 + 1] * (1.f / NG) - mu * mu;
      float rs = rsqrtf(var + EPS);
      for (int j = 0; j < 128; ++j) {
        int n = l + (j << 4);
        float val = zbuf[mrow * 2052 + n];
        zbuf[mrow * 2052 + n] = (val - mu) * rs * kn_g[n] + kn_b[n];
      }
    }
    __syncthreads();

    // ---- Phase B: y = h @ rec_kernel via bf16 WMMA, f32 accumulate ----
    v8f acc[16] = {};
    for (int kt = 0; kt < 16; ++kt) {
      ABfrag a;   // A-fragment: row m = l, dword v holds k = hi*16 + 2v, +1 (contiguous, same as B)
      #pragma unroll
      for (int v = 0; v < 8; ++v) {
        int dwi = kt * 16 + hi * 8 + v;
        a.u[v] = hbf[l * 260 + dwi];
      }
      #pragma unroll
      for (int nt = 0; nt < 16; ++nt) {
        int f = kt * 128 + waveId * 16 + nt;
        ABfrag b;
        b.q[0] = bfrag[(f * 2 + 0) * 32 + lane];
        b.q[1] = bfrag[(f * 2 + 1) * 32 + lane];
        acc[nt] = __builtin_amdgcn_wmma_f32_16x16x32_bf16(
            false, a.v, false, b.v, (short)0, acc[nt], false, false);
      }
    }
    // per-row stats of y (C/D layout: VGPR v holds row v + hi*8, n = l)
    #pragma unroll
    for (int v = 0; v < 8; ++v) {
      float s = 0.f, sq = 0.f;
      #pragma unroll
      for (int nt = 0; nt < 16; ++nt) { float y = acc[nt][v]; s += y; sq += y * y; }
      s = red16(s); sq = red16(sq);
      if (l == 0) {
        int r = v + hi * 8;
        atomicAdd(&yred[r * 2], s);
        atomicAdd(&yred[r * 2 + 1], sq);
      }
    }
    __syncthreads();
    // z = LN(zx) + LN(y)*rn_g + rn_b + bias  (write back into zbuf)
    #pragma unroll
    for (int v = 0; v < 8; ++v) {
      int r = v + hi * 8;
      float mu = yred[r * 2] * (1.f / NG);
      float var = yred[r * 2 + 1] * (1.f / NG) - mu * mu;
      float rs = rsqrtf(var + EPS);
      #pragma unroll
      for (int nt = 0; nt < 16; ++nt) {
        int col = waveId * 256 + nt * 16 + l;
        float yn = (acc[nt][v] - mu) * rs;
        zbuf[r * 2052 + col] += yn * rn_g[col] + rn_b[col] + bias[col];
      }
    }
    __syncthreads();

    // ---- Gate phase pass 1: c = sig(zf)*c + sig(zi)*tanh(zg), stats ----
    float zo_save[32];
    float cs = 0.f, csq = 0.f;
    #pragma unroll
    for (int j = 0; j < 32; ++j) {
      int u = l + (j << 4);
      float zi = zbuf[mrow * 2052 + u];
      float zf = zbuf[mrow * 2052 + 512 + u];
      float zg = zbuf[mrow * 2052 + 1024 + u];
      float zo = zbuf[mrow * 2052 + 1536 + u];
      float c  = sigf(zf) * cL[mrow * 516 + u] + sigf(zi) * tanhf(zg);
      cL[mrow * 516 + u] = c;
      cs += c; csq += c * c;
      zo_save[j] = zo;
    }
    cs = red16(cs); csq = red16(csq);
    if (l == 0) { cred[mrow * 2] = cs; cred[mrow * 2 + 1] = csq; }
    __syncthreads();
    // ---- pass 2: c = LN(c); h = sig(zo)*tanh(c); Dense(4)+relu ----
    {
      float mu = cred[mrow * 2] * (1.f / USZ);
      float var = cred[mrow * 2 + 1] * (1.f / USZ) - mu * mu;
      float rs = rsqrtf(var + EPS);
      float4 pd = make_float4(0.f, 0.f, 0.f, 0.f);
      #pragma unroll
      for (int j = 0; j < 32; ++j) {
        int u = l + (j << 4);
        float c  = cL[mrow * 516 + u];
        float cn = (c - mu) * rs * sn_g[u] + sn_b[u];
        cL[mrow * 516 + u] = cn;
        float h = sigf(zo_save[j]) * tanhf(cn);
        hL[mrow * 516 + u] = h;
        const float4 w = *(const float4*)(Wd + u * 4);
        pd.x += h * w.x; pd.y += h * w.y; pd.z += h * w.z; pd.w += h * w.w;
      }
      pd.x = red16(pd.x); pd.y = red16(pd.y);
      pd.z = red16(pd.z); pd.w = red16(pd.w);
      if (l == 0) {
        float4 o;
        o.x = fmaxf(pd.x + bdv.x, 0.f);
        o.y = fmaxf(pd.y + bdv.y, 0.f);
        o.z = fmaxf(pd.z + bdv.z, 0.f);
        o.w = fmaxf(pd.w + bdv.w, 0.f);
        *(float4*)(out + ((b0 + mrow) * TSZ + t) * 4) = o;
      }
    }
    __syncthreads();
    // ---- repack h as bf16 A-source for next step ----
    for (int i = tid; i < ROWS * 256; i += 256) {
      int m = i >> 8, kd = i & 255;
      hbf[m * 260 + kd] = packbf(hL[m * 516 + 2 * kd], hL[m * 516 + 2 * kd + 1]);
    }
    __syncthreads();
  }
}

extern "C" void kernel_launch(void* const* d_in, const int* in_sizes, int n_in,
                              void* d_out, int out_size, void* d_ws, size_t ws_size,
                              hipStream_t stream) {
  const float* x    = (const float*)d_in[0];
  const float* h0   = (const float*)d_in[1];
  const float* c0   = (const float*)d_in[2];
  // d_in[3] = mask: all-ones in reference and unused by the step fn
  const float* kern = (const float*)d_in[4];
  const float* rk   = (const float*)d_in[5];
  const float* bias = (const float*)d_in[6];
  const float* kn_g = (const float*)d_in[7];
  const float* kn_b = (const float*)d_in[8];
  const float* rn_g = (const float*)d_in[9];
  const float* rn_b = (const float*)d_in[10];
  const float* sn_g = (const float*)d_in[11];
  const float* sn_b = (const float*)d_in[12];
  const float* Wd   = (const float*)d_in[13];
  const float* bd   = (const float*)d_in[14];

  // 1) pack rec_kernel into bf16 WMMA B-fragments (2 MB in d_ws)
  prep_pack_B<<<2048, 256, 0, stream>>>(rk, (unsigned*)d_ws);

  // 2) persistent per-batch-tile scan; dynamic LDS = 247,424 B (< 320 KB/WGP)
  const size_t ldsBytes =
      (size_t)(4 * NG + 16 * 2052 + 16 * 516 * 2 + 16 * 260 + 64 + 96) * 4;
  decoder_scan<<<128, 256, ldsBytes, stream>>>(
      x, h0, c0, kern, bias, kn_g, kn_b, rn_g, rn_b, sn_g, sn_b, Wd, bd,
      (const uint4*)d_ws, (float*)d_out);
}
